// att0_16_104_42984032698946
// MI455X (gfx1250) — compile-verified, matrix operand order checked by reading, not run on hardware
//
#include <hip/hip_runtime.h>
#include <hip/hip_bf16.h>

// ---- types for WMMA fragments ----
typedef _Float16 v16h_t __attribute__((ext_vector_type(16)));
typedef _Float16 h8_t   __attribute__((ext_vector_type(8)));
typedef float    v8f_t  __attribute__((ext_vector_type(8)));

#define Bn  4096
#define NI  20
#define Dd  64
#define AH_STRIDE 72   // halves; 144B rows -> conflict-free b128 LDS reads
#define F_STRIDE  68   // floats; padded f32 rows

// Load one lane's 16-half WMMA fragment: 16-bit A/B 16x32 layout (ISA 7.12.2):
// halves 0..7 = K kb..kb+7, halves 8..15 = K kb+16..kb+23, from row `row`.
__device__ __forceinline__ v16h_t load_frag(const _Float16* __restrict__ base,
                                            int stride, int row, int kb) {
    v16h_t f;
#pragma unroll
    for (int c = 0; c < 2; ++c) {
        const h8_t v = *(const h8_t*)(base + row * stride + kb + c * 16);
#pragma unroll
        for (int t = 0; t < 8; ++t) f[c * 8 + t] = v[t];
    }
    return f;
}

// One 16x16 output tile of  C(32x64) = A(32x64,f16,LDS) @ W(64x64,f16,global)^T
__device__ __forceinline__ v8f_t wmma_tile(const _Float16* __restrict__ sA,
                                           const _Float16* __restrict__ Wh,
                                           int lane, int mi, int ni) {
    v8f_t acc = {};
    const int l15  = lane & 15;
    const int rowA = mi * 16 + l15;      // A-matrix row this lane supplies
    const int e    = ni * 16 + l15;      // W row == output column this lane supplies
    const int koff = (lane & 16) ? 8 : 0;
#pragma unroll
    for (int ks = 0; ks < 2; ++ks) {
        const int kb = ks * 32 + koff;
        const v16h_t a   = load_frag(sA, AH_STRIDE, rowA, kb);
        const v16h_t bfr = load_frag(Wh, 64, e, kb);
        acc = __builtin_amdgcn_wmma_f32_16x16x32_f16(
            /*neg_a=*/false, a, /*neg_b=*/false, bfr,
            /*c_mod=*/(short)0, acc, /*reuse_a=*/false, /*reuse_b=*/false);
    }
    return acc;
}

// One-time (per launch) f32 -> f16 conversion of the three 64x64 weight mats.
__global__ __launch_bounds__(256) void cvt_weights_kernel(
    const float* __restrict__ a, const float* __restrict__ b,
    const float* __restrict__ c, _Float16* __restrict__ outh) {
    const int t = blockIdx.x * 256 + threadIdx.x;
    if (t < 4096)       outh[t] = (_Float16)a[t];
    else if (t < 8192)  outh[t] = (_Float16)b[t - 4096];
    else if (t < 12288) outh[t] = (_Float16)c[t - 8192];
}

__global__ __launch_bounds__(256) void afn_fused_kernel(
    const float* __restrict__ emb,      // (B, 21, 64)
    const _Float16* __restrict__ wh,    // [att_w | w1 | w2] f16, 3*4096
    const float* __restrict__ att_wb,   // (64)
    const float* __restrict__ att_h,    // (1, 64)
    const float* __restrict__ att_hb,   // (1)
    const float* __restrict__ w1b,      // (64)
    const float* __restrict__ w2b,      // (64)
    float* __restrict__ out)            // (B, 20, 64)
{
    __shared__ __align__(16) _Float16 sAh[32 * AH_STRIDE]; // f16 A operand (i_em, later att_value)
    __shared__ __align__(16) _Float16 sXh[32 * AH_STRIDE]; // f16 A operand (out1 + i_em)
    __shared__ __align__(16) float sI[NI * F_STRIDE];      // i_em f32
    __shared__ __align__(16) float sT[32 * F_STRIDE];      // t = i_em @ att_w^T
    __shared__ __align__(16) float sS[NI * 24];            // scores -> alphas
    __shared__ float sU[64], sWb[64], sW1b[64], sW2b[64], sHv[64];

    const int b    = blockIdx.x;
    const int tid  = threadIdx.x;
    const int lane = tid & 31;
    const int wid  = tid >> 5;          // 0..7
    const int mi   = wid & 1;
    const int ni   = wid >> 1;
    const float hb = att_hb[0];
    const _Float16* att_w_h = wh;
    const _Float16* w1_h    = wh + 4096;
    const _Float16* w2_h    = wh + 8192;

    // ---------------- phase 0: stage inputs ----------------
    for (int k = tid; k < 32 * AH_STRIDE; k += 256) sAh[k] = (_Float16)0.f;
    if (tid < 64) {
        sU[tid]   = emb[b * 21 * 64 + tid];   // u_em
        sWb[tid]  = att_wb[tid];
        sW1b[tid] = w1b[tid];
        sW2b[tid] = w2b[tid];
        sHv[tid]  = att_h[tid];
    }
    for (int k = tid; k < NI * 64; k += 256) {
        const int m = k >> 6, d = k & 63;
        const float v = emb[b * 21 * 64 + 64 + k];
        sI[m * F_STRIDE + d]   = v;
        sAh[m * AH_STRIDE + d] = (_Float16)v;
    }
    __syncthreads();

    // ---------------- phase 1: t = i_em @ att_w^T (no bias yet) ----------------
    {
        v8f_t t = wmma_tile(sAh, att_w_h, lane, mi, ni);
        const int e  = ni * 16 + (lane & 15);
        const int mb = mi * 16 + ((lane & 16) ? 8 : 0);
#pragma unroll
        for (int vg = 0; vg < 8; ++vg) sT[(mb + vg) * F_STRIDE + e] = t[vg];
    }
    __syncthreads();

    // ------- phase 2: pairwise scores; symmetric -> 210 unordered pairs -------
    // Rectangle enumeration (no sqrt): p<200: k=p/20, i=p%20, j=(i+k)%20
    //                                  p>=200: i=p-200 (0..9), j=i+10
    if (tid < 210) {
        const int p = tid;
        int i, j;
        if (p < 200) {
            const int k = p / 20;        // uniform per 20-lane group; magic-mul
            i = p - k * 20;
            j = i + k;
            if (j >= NI) j -= NI;
        } else {
            i = p - 200;
            j = i + 10;
        }
        const float* ti = sT + i * F_STRIDE;
        const float* tj = sT + j * F_STRIDE;
        float s = hb;
#pragma unroll 8
        for (int e = 0; e < 64; ++e) {
            float v = ti[e] + tj[e] + sWb[e];
            v = v > 0.f ? v : 0.01f * v;          // leaky_relu
            s = fmaf(sHv[e], v, s);
        }
        sS[i * 24 + j] = s;
        if (i != j) sS[j * 24 + i] = s;           // scores are symmetric
    }
    __syncthreads();

    // ---------------- phase 3: softmax over j ----------------
    if (tid < NI) {
        float mx = -3.4e38f;
#pragma unroll
        for (int j = 0; j < NI; ++j) mx = fmaxf(mx, sS[tid * 24 + j]);
        float sum = 0.f;
#pragma unroll
        for (int j = 0; j < NI; ++j) {
            const float ev = __expf(sS[tid * 24 + j] - mx);
            sS[tid * 24 + j] = ev;
            sum += ev;
        }
        const float inv = 1.f / sum;
#pragma unroll
        for (int j = 0; j < NI; ++j) sS[tid * 24 + j] *= inv;
    }
    __syncthreads();

    // ---------------- phase 4: att_value = alphas @ i_em (into f16 A buffer) ----------------
    for (int o = tid; o < NI * 64; o += 256) {
        const int i = o >> 6, d = o & 63;
        float acc = 0.f;
#pragma unroll
        for (int j = 0; j < NI; ++j) acc = fmaf(sS[i * 24 + j], sI[j * F_STRIDE + d], acc);
        sAh[i * AH_STRIDE + d] = (_Float16)acc;   // rows 20..31 stay zero
    }
    __syncthreads();

    // ---------------- phase 5: out1 = att_value @ w1^T + w1b;  x = out1 + i_em ----------------
    {
        v8f_t o1 = wmma_tile(sAh, w1_h, lane, mi, ni);
        const int e  = ni * 16 + (lane & 15);
        const int mb = mi * 16 + ((lane & 16) ? 8 : 0);
#pragma unroll
        for (int vg = 0; vg < 8; ++vg) {
            const int m = mb + vg;
            const float x = (m < NI) ? (o1[vg] + sW1b[e] + sI[m * F_STRIDE + e]) : 0.f;
            sXh[m * AH_STRIDE + e] = (_Float16)x;
        }
    }
    __syncthreads();

    // ---------------- phase 6: att_ii = x @ w2^T + w2b; em_out = leaky(ui + att_ii) ----------------
    {
        v8f_t o2 = wmma_tile(sXh, w2_h, lane, mi, ni);
        const int e  = ni * 16 + (lane & 15);
        const int mb = mi * 16 + ((lane & 16) ? 8 : 0);
        float* outb = out + (size_t)b * NI * 64;
#pragma unroll
        for (int vg = 0; vg < 8; ++vg) {
            const int m = mb + vg;
            if (m < NI) {
                float r = fmaf(sU[e], sI[m * F_STRIDE + e], o2[vg] + sW2b[e]);
                r = r > 0.f ? r : 0.01f * r;      // final leaky_relu
                outb[m * 64 + e] = r;
            }
        }
    }
}

extern "C" void kernel_launch(void* const* d_in, const int* in_sizes, int n_in,
                              void* d_out, int out_size, void* d_ws, size_t ws_size,
                              hipStream_t stream) {
    const float* emb    = (const float*)d_in[0];
    const float* att_w  = (const float*)d_in[1];
    const float* att_wb = (const float*)d_in[2];
    const float* att_h  = (const float*)d_in[3];
    const float* att_hb = (const float*)d_in[4];
    const float* w1     = (const float*)d_in[5];
    const float* w1b    = (const float*)d_in[6];
    const float* w2     = (const float*)d_in[7];
    const float* w2b    = (const float*)d_in[8];
    float* out = (float*)d_out;

    _Float16* wh = (_Float16*)d_ws;     // 3 * 4096 halves = 24 KB of scratch
    cvt_weights_kernel<<<dim3(48), dim3(256), 0, stream>>>(att_w, w1, w2, wh);
    afn_fused_kernel<<<dim3(Bn), dim3(256), 0, stream>>>(
        emb, wh, att_wb, att_h, att_hb, w1b, w2b, out);
}
